// GNN_54511724921373
// MI455X (gfx1250) — compile-verified, matrix operand order checked by reading, not run on hardware
//
#include <hip/hip_runtime.h>
#include <math.h>

#define N_AG 4096
#define FDIM 512
#define PDIM 512
#define ADIM 512
#define SDIM 6
#define HDIM 20
#define GDIM 2048   // 4*FDIM (lstm gates)
#define KIN0 1024   // ADIM + FDIM

typedef __attribute__((ext_vector_type(16))) __bf16 v16bf;
typedef __attribute__((ext_vector_type(8)))  float  v8f;
typedef int v4i __attribute__((ext_vector_type(4)));

// pointer-to-int4 in global (AS1) / LDS (AS3) address spaces, as required by
// __builtin_amdgcn_global_load_async_to_lds_b128
typedef __attribute__((address_space(1))) v4i* gv4p;
typedef __attribute__((address_space(3))) v4i* lv4p;

union Frag16 { v16bf v; unsigned int u[8]; };

#if __has_builtin(__builtin_amdgcn_global_load_async_to_lds_b128)
#define USE_ASYNC_LDS 1
#endif

__device__ __forceinline__ unsigned short f2bf(float f) {
  unsigned int u = __float_as_uint(f);
  unsigned int r = u + 0x7fffu + ((u >> 16) & 1u);   // round-to-nearest-even
  return (unsigned short)(r >> 16);
}
__device__ __forceinline__ float bf2f(unsigned short s) {
  return __uint_as_float(((unsigned int)s) << 16);
}

// 16-byte global->LDS copy: async DMA on CDNA5, plain copy fallback
__device__ __forceinline__ void copy16(unsigned short* l, const unsigned short* g) {
#if defined(USE_ASYNC_LDS)
  __builtin_amdgcn_global_load_async_to_lds_b128(
      (gv4p)(size_t)(const void*)g,
      (lv4p)(unsigned)(size_t)(void*)l, 0, 0);
#else
  *(uint4*)l = *(const uint4*)g;
#endif
}
__device__ __forceinline__ void wait_async() {
#if defined(USE_ASYNC_LDS)
#if __has_builtin(__builtin_amdgcn_s_wait_asynccnt)
  __builtin_amdgcn_s_wait_asynccnt(0);
#else
  asm volatile("s_wait_asynccnt 0x0" ::: "memory");
#endif
#endif
}

// build a 16x32 bf16 fragment from one LDS tile row block (row stride 64B)
__device__ __forceinline__ void loadFrag(Frag16& f, const unsigned short* r, int hlf) {
  const uint4* p0 = (const uint4*)(r + hlf * 8);        // K pairs 0..7 (+8 for hi half)
  const uint4* p1 = (const uint4*)(r + 16 + hlf * 8);   // K pairs 16..23 (+8)
  uint4 x = *p0, y = *p1;
  f.u[0] = x.x; f.u[1] = x.y; f.u[2] = x.z; f.u[3] = x.w;
  f.u[4] = y.x; f.u[5] = y.y; f.u[6] = y.z; f.u[7] = y.w;
}

// ---------------------------------------------------------------------------
// WMMA GEMM (bf16 inputs): C[M,ldc] = act(A[M,K] @ WT[N,K]^T + bias)
// block = 256 thr (8 waves); tile 128(M) x 64(N); K step 32; double-buffered
// async LDS staging. Each wave: 32x32 patch = 4 WMMAs / K-step.
// ---------------------------------------------------------------------------
__global__ __launch_bounds__(256)
void wmma_gemm_bf16(const unsigned short* __restrict__ A,   // [M,K] bf16
                    const unsigned short* __restrict__ WT,  // [N,K] bf16
                    int K,
                    const float* __restrict__ bias,
                    float* __restrict__ C, int ldc, int applyRelu)
{
  __shared__ unsigned short ldsA[2][128][32];
  __shared__ unsigned short ldsB[2][64][32];
  const int tid  = threadIdx.x;
  const int lane = tid & 31;
  const int wave = tid >> 5;
  const int mw = wave & 3;                  // M strip (32 rows)
  const int nw = wave >> 2;                 // N strip (32 cols)
  const int m0 = blockIdx.y * 128;
  const int n0 = blockIdx.x * 64;
  const int hlf = lane >> 4;
  const int l15 = lane & 15;

  v8f acc00 = {}, acc01 = {}, acc10 = {}, acc11 = {};

  auto stage = [&](int buf, int k0) {
    // 768 chunks of 16B: 512 for A (128 rows x 4), 256 for B (64 rows x 4)
    for (int c = tid; c < 768; c += 256) {
      if (c < 512) {
        int row = c >> 2, ch = c & 3;
        copy16(&ldsA[buf][row][ch * 8], A + (size_t)(m0 + row) * K + k0 + ch * 8);
      } else {
        int cc = c - 512;
        int row = cc >> 2, ch = cc & 3;
        copy16(&ldsB[buf][row][ch * 8], WT + (size_t)(n0 + row) * K + k0 + ch * 8);
      }
    }
  };

  stage(0, 0);
  wait_async();
  __syncthreads();

  const int nsteps = K >> 5;
  for (int s = 0; s < nsteps; ++s) {
    const int cur = s & 1;
    if (s + 1 < nsteps) stage(cur ^ 1, (s + 1) << 5);

    Frag16 a0, a1, b0, b1;
    loadFrag(a0, &ldsA[cur][mw * 32 + l15][0], hlf);
    loadFrag(a1, &ldsA[cur][mw * 32 + 16 + l15][0], hlf);
    loadFrag(b0, &ldsB[cur][nw * 32 + l15][0], hlf);
    loadFrag(b1, &ldsB[cur][nw * 32 + 16 + l15][0], hlf);

    acc00 = __builtin_amdgcn_wmma_f32_16x16x32_bf16(false, a0.v, false, b0.v, (short)0, acc00, false, false);
    acc01 = __builtin_amdgcn_wmma_f32_16x16x32_bf16(false, a0.v, false, b1.v, (short)0, acc01, false, false);
    acc10 = __builtin_amdgcn_wmma_f32_16x16x32_bf16(false, a1.v, false, b0.v, (short)0, acc10, false, false);
    acc11 = __builtin_amdgcn_wmma_f32_16x16x32_bf16(false, a1.v, false, b1.v, (short)0, acc11, false, false);

    wait_async();
    __syncthreads();
  }

  const int colA = n0 + nw * 32 + l15;
  const int colB = colA + 16;
  const float bA = bias ? bias[colA] : 0.f;
  const float bB = bias ? bias[colB] : 0.f;
  #pragma unroll
  for (int r = 0; r < 8; ++r) {
    int row0 = m0 + mw * 32 + hlf * 8 + r;   // C layout: VGPR r -> M = r + 8*laneHalf
    int row1 = row0 + 16;
    float v00 = acc00[r] + bA, v01 = acc01[r] + bB;
    float v10 = acc10[r] + bA, v11 = acc11[r] + bB;
    if (applyRelu) {
      v00 = fmaxf(v00, 0.f); v01 = fmaxf(v01, 0.f);
      v10 = fmaxf(v10, 0.f); v11 = fmaxf(v11, 0.f);
    }
    C[(size_t)row0 * ldc + colA] = v00;
    C[(size_t)row0 * ldc + colB] = v01;
    C[(size_t)row1 * ldc + colA] = v10;
    C[(size_t)row1 * ldc + colB] = v11;
  }
}

// f32 [n] -> bf16 [n]
__global__ void convert_bf16(const float* __restrict__ in, unsigned short* __restrict__ out, int n) {
  int i = blockIdx.x * blockDim.x + threadIdx.x;
  if (i < n) out[i] = f2bf(in[i]);
}

// W[K,N] f32 -> WT[N,K] bf16 (LDS-tiled transpose)
__global__ __launch_bounds__(256)
void transpose_to_bf16(const float* __restrict__ in, unsigned short* __restrict__ out,
                       int K, int N)
{
  __shared__ float tile[32][33];
  const int nb = blockIdx.x * 32, kb = blockIdx.y * 32;
  const int tx = threadIdx.x & 31, ty = threadIdx.x >> 5;  // 8 rows/pass
  for (int r = ty; r < 32; r += 8) tile[r][tx] = in[(size_t)(kb + r) * N + nb + tx];
  __syncthreads();
  for (int r = ty; r < 32; r += 8) out[(size_t)(nb + r) * K + kb + tx] = f2bf(tile[tx][r]);
}

// ---------------------------------------------------------------------------
// phi MLP: msg = [x[src], states[src]-states[dst]] ; 518->20->20->512
// ---------------------------------------------------------------------------
__global__ __launch_bounds__(256)
void phi_mlp_kernel(const float* __restrict__ xin, const float* __restrict__ states,
                    const int* __restrict__ src, const int* __restrict__ dst,
                    const float* __restrict__ w0, const float* __restrict__ pb0,
                    const float* __restrict__ w1, const float* __restrict__ pb1,
                    const float* __restrict__ w2, const float* __restrict__ pb2,
                    float* __restrict__ phi)
{
  __shared__ float msg[FDIM + SDIM];
  __shared__ float h0[HDIM], h1[HDIM];
  const int e = blockIdx.x;
  const int sE = src[e], dE = dst[e];
  const int tid = threadIdx.x;
  for (int c = tid; c < FDIM; c += 256) msg[c] = xin[(size_t)sE * FDIM + c];
  if (tid < SDIM)
    msg[FDIM + tid] = states[(size_t)sE * SDIM + tid] - states[(size_t)dE * SDIM + tid];
  __syncthreads();
  if (tid < HDIM) {
    float acc = pb0[tid];
    for (int c = 0; c < FDIM + SDIM; ++c) acc += msg[c] * w0[c * HDIM + tid];
    h0[tid] = fmaxf(acc, 0.f);
  }
  __syncthreads();
  if (tid < HDIM) {
    float acc = pb1[tid];
    for (int j = 0; j < HDIM; ++j) acc += h0[j] * w1[j * HDIM + tid];
    h1[tid] = fmaxf(acc, 0.f);
  }
  __syncthreads();
  for (int c = tid; c < PDIM; c += 256) {
    float acc = pb2[c];
    for (int j = 0; j < HDIM; ++j) acc += h1[j] * w2[j * PDIM + c];
    phi[(size_t)e * PDIM + c] = acc;
  }
}

// ---------------------------------------------------------------------------
// attention: general segment softmax over dst (atomics)
// ---------------------------------------------------------------------------
__device__ __forceinline__ unsigned int fkey(float f) {
  unsigned int u = __float_as_uint(f);
  return (u & 0x80000000u) ? ~u : (u | 0x80000000u);
}
__device__ __forceinline__ float funkey(unsigned int k) {
  unsigned int u = (k & 0x80000000u) ? (k & 0x7fffffffu) : ~k;
  return __uint_as_float(u);
}

__global__ void attn_init(unsigned int* mkey, float* ssum) {
  int i = blockIdx.x * blockDim.x + threadIdx.x;
  if (i < N_AG) { mkey[i] = 0u; ssum[i] = 0.f; }
}

__global__ __launch_bounds__(128)
void attn_logits(const float* __restrict__ q, const float* __restrict__ k,
                 const int* __restrict__ src, const int* __restrict__ dst,
                 float* __restrict__ logit, unsigned int* __restrict__ mkey)
{
  const int e = blockIdx.x * 4 + (threadIdx.x >> 5);
  const int lane = threadIdx.x & 31;
  const float* qr = q + (size_t)dst[e] * ADIM;
  const float* kr = k + (size_t)src[e] * ADIM;
  float s = 0.f;
  for (int c = lane; c < ADIM; c += 32) s += qr[c] * kr[c];
  for (int off = 16; off; off >>= 1) s += __shfl_xor(s, off, 32);
  if (lane == 0) {
    s *= 0.044194173824159216f;  // 1/sqrt(512)
    logit[e] = s;
    atomicMax(mkey + dst[e], fkey(s));
  }
}

__global__ void attn_expsum(const float* __restrict__ logit,
                            const unsigned int* __restrict__ mkey,
                            const int* __restrict__ dst,
                            float* __restrict__ expe, float* __restrict__ ssum)
{
  int e = blockIdx.x * blockDim.x + threadIdx.x;
  if (e < N_AG) {
    float m = funkey(mkey[dst[e]]);
    float ee = __expf(logit[e] - m);
    expe[e] = ee;
    atomicAdd(ssum + dst[e], ee);
  }
}

__global__ __launch_bounds__(256)
void attn_aggr(const float* __restrict__ expe, const float* __restrict__ ssum,
               const float* __restrict__ v, const int* __restrict__ src,
               const int* __restrict__ dst, float* __restrict__ aggr)
{
  const int e = blockIdx.x;
  const float alpha = expe[e] / ssum[dst[e]];
  const float* vr = v + (size_t)src[e] * ADIM;
  float* ar = aggr + (size_t)dst[e] * ADIM;
  for (int c = threadIdx.x; c < ADIM; c += 256) atomicAdd(ar + c, alpha * vr[c]);
}

// gi = [aggr | xin]  (N x 1024)
__global__ void concat_gi(const float* __restrict__ aggr, const float* __restrict__ xin,
                          float* __restrict__ gi)
{
  int idx = blockIdx.x * blockDim.x + threadIdx.x;
  int row = idx >> 10, col = idx & 1023;
  if (row < N_AG)
    gi[idx] = (col < ADIM) ? aggr[(size_t)row * ADIM + col]
                           : xin[(size_t)row * FDIM + (col - ADIM)];
}

// whh[512][2048] f32 -> whhT[2048][512] bf16
__global__ void prep_whhT(const float* __restrict__ whh, unsigned short* __restrict__ whhT) {
  int idx = blockIdx.x * blockDim.x + threadIdx.x;
  if (idx < GDIM * FDIM) {
    int c = idx / FDIM, kk = idx % FDIM;
    whhT[idx] = f2bf(whh[(size_t)kk * GDIM + c]);
  }
}

// ---------------------------------------------------------------------------
// Sequential LSTM scan over 4096 steps. 16 blocks x 512 thr.
// Block b owns h[b*32 .. b*32+32); double-buffered global h exchange +
// monotonic atomic-counter device barrier per step.
// ---------------------------------------------------------------------------
__global__ __launch_bounds__(512)
void lstm_scan(const float* __restrict__ pre, const unsigned short* __restrict__ whhT,
               float* __restrict__ hout, float* __restrict__ hbuf,
               unsigned int* __restrict__ cnt)
{
  __shared__ float hsh[FDIM];
  __shared__ float psum[512];
  __shared__ float gbuf[128];
  __shared__ float cst[32];
  const int tid = threadIdx.x;
  const int b = blockIdx.x;                       // 0..15
  const int colLocal = tid >> 2;                  // 0..127
  const int part = tid & 3;                       // K quarter
  const int jo = colLocal & 31;
  const int gtype = colLocal >> 5;                // 0=i 1=f 2=g 3=o
  const int j0 = b * 32;
  const int gcol = gtype * FDIM + j0 + jo;
  const unsigned int* w32 = (const unsigned int*)(whhT + (size_t)gcol * FDIM);

  if (tid < FDIM) hsh[tid] = 0.f;
  if (tid < 32) cst[tid] = 0.f;
  __syncthreads();

  for (int t = 0; t < N_AG; ++t) {
    float acc = (part == 0) ? pre[(size_t)t * GDIM + gcol] : 0.f;
    const int ks = part * 128;
    #pragma unroll 8
    for (int kk = 0; kk < 64; ++kk) {
      unsigned int u = w32[(ks >> 1) + kk];
      acc += hsh[ks + 2 * kk]     * bf2f((unsigned short)(u & 0xffffu));
      acc += hsh[ks + 2 * kk + 1] * bf2f((unsigned short)(u >> 16));
    }
    psum[tid] = acc;
    __syncthreads();
    if (tid < 128)
      gbuf[tid] = psum[tid * 4] + psum[tid * 4 + 1] + psum[tid * 4 + 2] + psum[tid * 4 + 3];
    __syncthreads();
    float* hb = hbuf + (t & 1) * FDIM;
    if (tid < 32) {
      float gi_ = gbuf[tid], gf = gbuf[32 + tid], gg = gbuf[64 + tid], go = gbuf[96 + tid];
      float c = (1.f / (1.f + __expf(-gf))) * cst[tid]
              + (1.f / (1.f + __expf(-gi_))) * tanhf(gg);
      float h = (1.f / (1.f + __expf(-go))) * tanhf(c);
      cst[tid] = c;
      hout[(size_t)t * FDIM + j0 + tid] = h;
      hb[j0 + tid] = h;
    }
    __syncthreads();
    if (tid == 0) {
      __builtin_amdgcn_fence(__ATOMIC_RELEASE, "agent");
      __hip_atomic_fetch_add(cnt, 1u, __ATOMIC_RELAXED, __HIP_MEMORY_SCOPE_AGENT);
      const unsigned int target = 16u * (unsigned)(t + 1);
      while (__hip_atomic_load(cnt, __ATOMIC_RELAXED, __HIP_MEMORY_SCOPE_AGENT) < target)
        __builtin_amdgcn_s_sleep(1);
    }
    __syncthreads();
    __builtin_amdgcn_fence(__ATOMIC_ACQUIRE, "agent");
    if (tid < FDIM) hsh[tid] = hb[tid];
    __syncthreads();
  }
}

__global__ void relu_inplace(float* x, int n) {
  int i = blockIdx.x * blockDim.x + threadIdx.x;
  if (i < n) x[i] = fmaxf(x[i], 0.f);
}

// hcbf = relu(xcur) @ cbf_w + cbf_b : one wave per row
__global__ __launch_bounds__(256)
void cbf_kernel(const float* __restrict__ xcur, const float* __restrict__ w,
                const float* __restrict__ bias, float* __restrict__ out)
{
  const int row = blockIdx.x * 8 + (threadIdx.x >> 5);
  const int lane = threadIdx.x & 31;
  const float* xr = xcur + (size_t)row * FDIM;
  float s = 0.f;
  for (int c = lane; c < FDIM; c += 32) s += fmaxf(xr[c], 0.f) * w[c];
  for (int off = 16; off; off >>= 1) s += __shfl_xor(s, off, 32);
  if (lane == 0) out[row] = s + bias[0];
}

// ---------------------------------------------------------------------------
extern "C" void kernel_launch(void* const* d_in, const int* in_sizes, int n_in,
                              void* d_out, int out_size, void* d_ws, size_t ws_size,
                              hipStream_t stream)
{
  (void)in_sizes; (void)n_in; (void)out_size; (void)ws_size;
  const float* x      = (const float*)d_in[0];
  const float* states = (const float*)d_in[1];
  const int*   src    = (const int*)d_in[2];
  const int*   dst    = (const int*)d_in[3];
  const float* phi_w0 = (const float*)d_in[4];
  const float* phi_b0 = (const float*)d_in[5];
  const float* phi_w1 = (const float*)d_in[6];
  const float* phi_b1 = (const float*)d_in[7];
  const float* phi_w2 = (const float*)d_in[8];
  const float* phi_b2 = (const float*)d_in[9];
  const float* tq_w = (const float*)d_in[10]; const float* tq_b = (const float*)d_in[11];
  const float* tk_w = (const float*)d_in[12]; const float* tk_b = (const float*)d_in[13];
  const float* tv_w = (const float*)d_in[14]; const float* tv_b = (const float*)d_in[15];
  const float* ts_w = (const float*)d_in[16]; const float* ts_b = (const float*)d_in[17];
  const float* wih0 = (const float*)d_in[18]; const float* whh0 = (const float*)d_in[19];
  const float* lb0  = (const float*)d_in[20];
  const float* wih1 = (const float*)d_in[21]; const float* whh1 = (const float*)d_in[22];
  const float* lb1  = (const float*)d_in[23];
  const float* cbf_w = (const float*)d_in[24]; const float* cbf_b = (const float*)d_in[25];

  char* ws = (char*)d_ws;
  size_t off = 0;
  auto alloc = [&](size_t bytes) -> void* {
    void* p = ws + off;
    off = (off + bytes + 255) & ~(size_t)255;
    return p;
  };
  const size_t NF = (size_t)N_AG * FDIM * sizeof(float);     // 8 MB
  float* phi   = (float*)alloc(NF);
  float* q     = (float*)alloc(NF);
  float* k     = (float*)alloc(NF);
  float* v     = (float*)alloc(NF);
  float* sagg  = (float*)alloc(NF);
  float* gi    = (float*)alloc(2 * NF);
  float* pre   = (float*)alloc(4 * NF);                      // N x 2048
  float* h0seq = (float*)alloc(NF);
  float* bufA  = (float*)alloc(NF);
  float* bufB  = (float*)alloc(NF);
  unsigned short* whhT = (unsigned short*)alloc((size_t)GDIM * FDIM * sizeof(unsigned short));
  unsigned short* Abf  = (unsigned short*)alloc((size_t)N_AG * KIN0 * sizeof(unsigned short));
  unsigned short* WTbf = (unsigned short*)alloc((size_t)GDIM * KIN0 * sizeof(unsigned short));
  float* logit = (float*)alloc(N_AG * sizeof(float));
  float* expe  = (float*)alloc(N_AG * sizeof(float));
  float* ssum  = (float*)alloc(N_AG * sizeof(float));
  unsigned int* mkey = (unsigned int*)alloc(N_AG * sizeof(unsigned int));
  float* hbuf  = (float*)alloc(2 * FDIM * sizeof(float));
  unsigned int* cnt = (unsigned int*)alloc(256);

  const dim3 gProj(PDIM / 64, N_AG / 128);     // 4096 x 512 outputs
  const dim3 gPre(GDIM / 64, N_AG / 128);      // 4096 x 2048 outputs
  const dim3 gT512(PDIM / 32, PDIM / 32);      // transpose 512x512
  const dim3 gTih0(GDIM / 32, KIN0 / 32);      // transpose 1024x2048
  const dim3 gTih1(GDIM / 32, FDIM / 32);      // transpose 512x2048

  const float* xin = x;
  for (int l = 0; l < 4; ++l) {
    // 1) phi MLP, then convert phi to bf16 once (reused by 4 GEMMs)
    phi_mlp_kernel<<<N_AG, 256, 0, stream>>>(
        xin, states, src, dst,
        phi_w0 + (size_t)l * (FDIM + SDIM) * HDIM, phi_b0 + (size_t)l * HDIM,
        phi_w1 + (size_t)l * HDIM * HDIM,          phi_b1 + (size_t)l * HDIM,
        phi_w2 + (size_t)l * HDIM * PDIM,          phi_b2 + (size_t)l * PDIM, phi);
    convert_bf16<<<(N_AG * PDIM) / 256, 256, 0, stream>>>(phi, Abf, N_AG * PDIM);
    // 2) q,k,v,skip projections (WMMA, bf16)
    const float* tw[4] = { tq_w, tk_w, tv_w, ts_w };
    const float* tb[4] = { tq_b, tk_b, tv_b, ts_b };
    float* tout[4] = { q, k, v, sagg };
    for (int p = 0; p < 4; ++p) {
      transpose_to_bf16<<<gT512, 256, 0, stream>>>(tw[p] + (size_t)l * PDIM * ADIM, WTbf, PDIM, ADIM);
      wmma_gemm_bf16<<<gProj, 256, 0, stream>>>(Abf, WTbf, PDIM, tb[p] + (size_t)l * ADIM,
                                                tout[p], ADIM, 0);
    }
    // 3) segment softmax attention
    attn_init<<<N_AG / 256, 256, 0, stream>>>(mkey, ssum);
    attn_logits<<<N_AG / 4, 128, 0, stream>>>(q, k, src, dst, logit, mkey);
    attn_expsum<<<N_AG / 256, 256, 0, stream>>>(logit, mkey, dst, expe, ssum);
    attn_aggr<<<N_AG, 256, 0, stream>>>(expe, ssum, v, src, dst, sagg);
    // 4) LSTM layer 0: pre = [aggr|xin] @ wih0 + b0 (WMMA), then scan
    concat_gi<<<(N_AG * KIN0) / 256, 256, 0, stream>>>(sagg, xin, gi);
    convert_bf16<<<(N_AG * KIN0) / 256, 256, 0, stream>>>(gi, Abf, N_AG * KIN0);
    transpose_to_bf16<<<gTih0, 256, 0, stream>>>(wih0 + (size_t)l * KIN0 * GDIM, WTbf, KIN0, GDIM);
    wmma_gemm_bf16<<<gPre, 256, 0, stream>>>(Abf, WTbf, KIN0, lb0 + (size_t)l * GDIM,
                                             pre, GDIM, 0);
    prep_whhT<<<(GDIM * FDIM) / 256, 256, 0, stream>>>(whh0 + (size_t)l * FDIM * GDIM, whhT);
    (void)hipMemsetAsync(cnt, 0, sizeof(unsigned int), stream);
    lstm_scan<<<16, 512, 0, stream>>>(pre, whhT, h0seq, hbuf, cnt);
    // 5) LSTM layer 1
    convert_bf16<<<(N_AG * FDIM) / 256, 256, 0, stream>>>(h0seq, Abf, N_AG * FDIM);
    transpose_to_bf16<<<gTih1, 256, 0, stream>>>(wih1 + (size_t)l * FDIM * GDIM, WTbf, FDIM, GDIM);
    wmma_gemm_bf16<<<gPre, 256, 0, stream>>>(Abf, WTbf, FDIM, lb1 + (size_t)l * GDIM,
                                             pre, GDIM, 0);
    prep_whhT<<<(GDIM * FDIM) / 256, 256, 0, stream>>>(whh1 + (size_t)l * FDIM * GDIM, whhT);
    (void)hipMemsetAsync(cnt, 0, sizeof(unsigned int), stream);
    float* outX = (l == 3) ? (float*)d_out : ((l == 1) ? bufB : bufA);
    lstm_scan<<<16, 512, 0, stream>>>(pre, whhT, outX, hbuf, cnt);
    if (l == 0 || l == 2)
      relu_inplace<<<(N_AG * FDIM) / 256, 256, 0, stream>>>(outX, N_AG * FDIM);
    xin = outX;
  }
  // final CBF head
  cbf_kernel<<<N_AG / 8, 256, 0, stream>>>((const float*)d_out, cbf_w, cbf_b,
                                           (float*)d_out + (size_t)N_AG * FDIM);
}
